// AdaptiveMOELayer_7705171329369
// MI455X (gfx1250) — matrix operand order repaired and checked by reading, not compiled, not run on hardware
//
#include <hip/hip_runtime.h>
#include <math.h>

typedef _Float16 half_t;
typedef __attribute__((ext_vector_type(16))) _Float16 v16h;
typedef __attribute__((ext_vector_type(8)))  _Float16 v8h;
typedef __attribute__((ext_vector_type(8)))  float    v8f;

#define E_      8
#define KTOP    2
#define DMODEL  2048
#define DEXPERT 1024
#define DFF     1024
#define NTOK    8192           // B*T = 4*2048
#define CAP     2560           // ceil(NTOK*KTOP/E * 1.25)
#define NROUTE  (NTOK*KTOP)    // 16384

// ---------------------------------------------------------------- converts
__global__ void cvt_f16_kernel(half_t* __restrict__ dst, const float* __restrict__ src, int n) {
    int i = blockIdx.x * 256 + threadIdx.x;
    if (i < n) dst[i] = (half_t)src[i];
}

// src[b][r][c] (f32) -> dst[b][c][r] (f16)
__global__ void transpose_cvt_kernel(half_t* __restrict__ dst, const float* __restrict__ src,
                                     int Bn, int R, int C) {
    int i = blockIdx.x * 256 + threadIdx.x;
    int total = Bn * R * C;
    if (i >= total) return;
    int r = i % R;
    int t = i / R;
    int c = t % C;
    int b = t / C;
    dst[i] = (half_t)src[((size_t)b * R + r) * C + c];
}

__global__ void init_meta_kernel(int* __restrict__ slot_token, float* __restrict__ s_sum,
                                 int* __restrict__ counts) {
    int i = blockIdx.x * 256 + threadIdx.x;
    if (i < E_ * CAP) slot_token[i] = -1;
    if (i < E_) { s_sum[i] = 0.f; counts[i] = 0; }
}

// ---------------------------------------------------------------- router
__global__ __launch_bounds__(256)
void router_kernel(const float* __restrict__ X, const float* __restrict__ RW,
                   const float* __restrict__ GW,
                   int* __restrict__ topk_e, float* __restrict__ topk_g,
                   float* __restrict__ g_shared, float* __restrict__ s_sum,
                   int* __restrict__ counts) {
    __shared__ float red[9][256];
    const int n = blockIdx.x;
    const int tid = threadIdx.x;
    const float* x = X + (size_t)n * DMODEL;

    float acc[9];
#pragma unroll
    for (int j = 0; j < 9; ++j) acc[j] = 0.f;
    for (int d = tid; d < DMODEL; d += 256) {
        float xv = x[d];
#pragma unroll
        for (int j = 0; j < 8; ++j) acc[j] += xv * RW[j * DMODEL + d];
        acc[8] += xv * GW[d];
    }
#pragma unroll
    for (int j = 0; j < 9; ++j) red[j][tid] = acc[j];
    __syncthreads();
    for (int s = 128; s > 0; s >>= 1) {
        if (tid < s)
#pragma unroll
            for (int j = 0; j < 9; ++j) red[j][tid] += red[j][tid + s];
        __syncthreads();
    }
    if (tid == 0) {
        float lg[8], s8[8];
        float mx = -1e30f;
#pragma unroll
        for (int e = 0; e < 8; ++e) { lg[e] = red[e][0]; mx = fmaxf(mx, lg[e]); }
        float sum = 0.f;
#pragma unroll
        for (int e = 0; e < 8; ++e) { s8[e] = expf(lg[e] - mx); sum += s8[e]; }
        float inv = 1.f / sum;
        int i1 = 0; float v1 = -1.f;
#pragma unroll
        for (int e = 0; e < 8; ++e) {
            s8[e] *= inv;
            atomicAdd(&s_sum[e], s8[e]);
            if (s8[e] > v1) { v1 = s8[e]; i1 = e; }   // ties -> lowest index (jax top_k)
        }
        int i2 = 0; float v2 = -1.f;
#pragma unroll
        for (int e = 0; e < 8; ++e)
            if (e != i1 && s8[e] > v2) { v2 = s8[e]; i2 = e; }
        topk_e[n * 2 + 0] = i1; topk_g[n * 2 + 0] = v1;
        topk_e[n * 2 + 1] = i2; topk_g[n * 2 + 1] = v2;
        atomicAdd(&counts[i1], 1);
        atomicAdd(&counts[i2], 1);
        g_shared[n] = 1.f / (1.f + expf(-red[8][0]));
    }
}

// ------------------------------------------- ordered capacity routing scan
// Processes routes r = token*K + k in increasing r (== stable sort by expert),
// assigning per-expert slots; drops routes with position >= CAP.
__global__ __launch_bounds__(1024)
void route_scan_kernel(const int* __restrict__ topk_e, const float* __restrict__ topk_g,
                       int* __restrict__ slot_token, float* __restrict__ slot_gate) {
    __shared__ int cnt[32][8];
    __shared__ int base[8];
    const int tid = threadIdx.x;
    const int wave = tid >> 5;
    const int lane = tid & 31;
    if (tid < 8) base[tid] = 0;
    __syncthreads();

    for (int c = 0; c < NROUTE / 1024; ++c) {
        int r = c * 1024 + tid;
        int e = topk_e[r];
        unsigned bal[8];
#pragma unroll
        for (int ex = 0; ex < 8; ++ex) bal[ex] = __builtin_amdgcn_ballot_w32(e == ex);
        unsigned lt = (1u << lane) - 1u;
        int wrank = __popc(bal[e] & lt);
        if (lane < 8) cnt[wave][lane] = __popc(bal[lane]);
        __syncthreads();
        int pre = 0;
        for (int w = 0; w < wave; ++w) pre += cnt[w][e];
        int pos = base[e] + pre + wrank;
        if (pos < CAP) {
            slot_token[e * CAP + pos] = r >> 1;      // token id
            slot_gate[e * CAP + pos] = topk_g[r];
        }
        __syncthreads();
        if (tid < 8) {
            int tot = 0;
            for (int w = 0; w < 32; ++w) tot += cnt[w][tid];
            base[tid] += tot;
        }
        __syncthreads();
    }
}

// ---------------------------------------------------------------- WMMA GEMM
// C[M,N] = A[M,K] * BT[N,K]^T   (BT is K-contiguous per output column)
// Wave tile 64x64: 4 A-frags x 4 B-frags -> 16 v_wmma per K-step of 32.
// Block = 4 waves stacked in M -> block tile 256x64.
// MODE 0: exact GELU -> f16 store into Hout
// MODE 1: Yout[row,col] = row_gate[row] * acc        (f32, full overwrite)
// MODE 2: t = rmap[row]; if t>=0 atomicAdd(&Yout[t,col], gate[row]*acc)
// GATHER: A rows fetched via rmap (slot -> token into Abase); t<0 -> zeros.
__device__ __forceinline__ float gelu_exact(float x) {
    return 0.5f * x * (1.f + erff(x * 0.70710678118654752f));
}

#define PF_ELEMS 256   // prefetch distance: 8 K-steps ahead (512 bytes)

template <int MODE, bool GATHER>
__global__ __launch_bounds__(128)
void gemm_wmma_kernel(const half_t* __restrict__ Abase, const half_t* __restrict__ BTbase,
                      half_t* __restrict__ Hout, float* __restrict__ Yout,
                      const int* __restrict__ row_map, const float* __restrict__ row_gate,
                      int Kdim, int lda, int ldb, int ldh,
                      int Mper, int Aper, int Bper, int Hper) {
    const int e = blockIdx.z;
    const half_t* BT = BTbase + (size_t)e * Bper;
    const int* rmap = (row_map != nullptr) ? (row_map + (size_t)e * Mper) : nullptr;

    const int tid = threadIdx.x;
    const int wave = tid >> 5;
    const int lane = tid & 31;
    const int laneM = lane & 15;
    const int laneHi = lane >> 4;

    const int mBase = blockIdx.x * 256 + wave * 64;  // this wave: rows [mBase, mBase+64)
    const int nBase = blockIdx.y * 64;               // this block: cols [nBase, nBase+64)

    v8f zero8;
#pragma unroll
    for (int i = 0; i < 8; ++i) zero8[i] = 0.f;
    v8f acc[4][4];
#pragma unroll
    for (int mi = 0; mi < 4; ++mi)
#pragma unroll
        for (int ni = 0; ni < 4; ++ni) acc[mi][ni] = zero8;

    // per-lane A row pointers (A fragment: row M = laneM, K chunks at laneHi*16, +8)
    const half_t* aptr[4];
    bool avalid[4];
#pragma unroll
    for (int mi = 0; mi < 4; ++mi) {
        int m = mBase + mi * 16 + laneM;
        if (GATHER) {
            int t = rmap[m];
            avalid[mi] = (t >= 0);
            aptr[mi] = Abase + (size_t)(t < 0 ? 0 : t) * lda;
        } else {
            avalid[mi] = true;
            aptr[mi] = Abase + (size_t)e * Aper + (size_t)m * lda;
        }
    }
    // per-lane B column pointers (B fragment: col N = laneM, K = 16*laneHi + h)
    const half_t* bptr[4];
#pragma unroll
    for (int ni = 0; ni < 4; ++ni) {
        int ncol = nBase + ni * 16 + laneM;
        bptr[ni] = BT + (size_t)ncol * ldb;
    }
    const int kA = laneHi * 16;
    const int kB = laneHi * 16;

    for (int k0 = 0; k0 < Kdim; k0 += 32) {
        // speculative stream prefetch (global_prefetch_b8); overshoot is harmless
#pragma unroll
        for (int ni = 0; ni < 4; ++ni)
            __builtin_prefetch((const void*)(bptr[ni] + k0 + kB + PF_ELEMS), 0, 1);

        v16h a[4];
#pragma unroll
        for (int mi = 0; mi < 4; ++mi) {
            if (avalid[mi]) {
                v8h lo = *(const v8h*)(aptr[mi] + k0 + kA);
                v8h hi = *(const v8h*)(aptr[mi] + k0 + kA + 8);
#pragma unroll
                for (int i = 0; i < 8; ++i) { a[mi][i] = lo[i]; a[mi][8 + i] = hi[i]; }
            } else {
#pragma unroll
                for (int i = 0; i < 16; ++i) a[mi][i] = (half_t)0.f;
            }
        }
        v16h b[4];
#pragma unroll
        for (int ni = 0; ni < 4; ++ni) b[ni] = *(const v16h*)(bptr[ni] + k0 + kB);
#pragma unroll
        for (int mi = 0; mi < 4; ++mi)
#pragma unroll
            for (int ni = 0; ni < 4; ++ni)
                acc[mi][ni] = __builtin_amdgcn_wmma_f32_16x16x32_f16(
                    false, a[mi], false, b[ni], (short)0, acc[mi][ni], false, false);
    }

    // epilogue: C element (v, lane): row = v + 8*laneHi, col = laneM
#pragma unroll
    for (int mi = 0; mi < 4; ++mi) {
#pragma unroll
        for (int v = 0; v < 8; ++v) {
            int row = mBase + mi * 16 + v + laneHi * 8;
            int tok = 0;
            float g = 0.f;
            if (MODE == 2) {
                tok = rmap[row];
                g = (row_gate + (size_t)e * Mper)[row];
            } else if (MODE == 1) {
                g = row_gate[row];
            }
#pragma unroll
            for (int ni = 0; ni < 4; ++ni) {
                int col = nBase + ni * 16 + laneM;
                float val = acc[mi][ni][v];
                if (MODE == 0) {
                    Hout[(size_t)e * Hper + (size_t)row * ldh + col] = (half_t)gelu_exact(val);
                } else if (MODE == 1) {
                    Yout[(size_t)row * DMODEL + col] = g * val;
                } else {
                    if (tok >= 0)
                        atomicAdd(&Yout[(size_t)tok * DMODEL + col], g * val);
                }
            }
        }
    }
}

// ---------------------------------------------------------------- stats
__global__ void stats_kernel(float* __restrict__ tail, const float* __restrict__ s_sum,
                             const int* __restrict__ counts) {
    int i = threadIdx.x;
    if (i < E_) {
        tail[i] = s_sum[i] / (float)NTOK;                    // importance
        tail[E_ + i] = (float)counts[i] / (float)NROUTE;     // load
    }
}

// ---------------------------------------------------------------- launcher
extern "C" void kernel_launch(void* const* d_in, const int* in_sizes, int n_in,
                              void* d_out, int out_size, void* d_ws, size_t ws_size,
                              hipStream_t stream) {
    (void)in_sizes; (void)n_in; (void)out_size; (void)ws_size;
    const float* X   = (const float*)d_in[0];   // [4,2048,2048]
    const float* RW  = (const float*)d_in[1];   // [8,2048]
    const float* GW  = (const float*)d_in[2];   // [1,2048]
    const float* W1  = (const float*)d_in[3];   // [8,2048,1024]
    const float* W2  = (const float*)d_in[4];   // [8,1024,2048]
    const float* SW1 = (const float*)d_in[5];   // [1024,2048]
    const float* SW2 = (const float*)d_in[6];   // [2048,1024]
    float* y = (float*)d_out;                   // [8192*2048] + importance[8] + load[8]

    char* p = (char*)d_ws;
    auto alloc = [&](size_t bytes) -> char* {
        char* r = p;
        p += (bytes + 255) & ~(size_t)255;
        return r;
    };
    half_t* xh    = (half_t*)alloc((size_t)NTOK * DMODEL * 2);
    half_t* w1t   = (half_t*)alloc((size_t)E_ * DEXPERT * DMODEL * 2); // [e][f][d]
    half_t* w2t   = (half_t*)alloc((size_t)E_ * DMODEL * DEXPERT * 2); // [e][d][f]
    half_t* sw1h  = (half_t*)alloc((size_t)DFF * DMODEL * 2);          // [f][d]
    half_t* sw2h  = (half_t*)alloc((size_t)DMODEL * DFF * 2);          // [d][f]
    half_t* hmoe  = (half_t*)alloc((size_t)E_ * CAP * DEXPERT * 2);
    half_t* hsh   = (half_t*)alloc((size_t)NTOK * DFF * 2);
    float*  gsh   = (float*)alloc((size_t)NTOK * 4);
    int*    tke   = (int*)alloc((size_t)NROUTE * 4);
    float*  tkg   = (float*)alloc((size_t)NROUTE * 4);
    int*    stok  = (int*)alloc((size_t)E_ * CAP * 4);
    float*  sgate = (float*)alloc((size_t)E_ * CAP * 4);
    float*  ssum  = (float*)alloc(64);
    int*    cnts  = (int*)alloc(64);

    // 1) precision conversions / weight transposes
    {
        int n = NTOK * DMODEL;
        cvt_f16_kernel<<<(n + 255) / 256, 256, 0, stream>>>(xh, X, n);
        int tw = E_ * DMODEL * DEXPERT;
        transpose_cvt_kernel<<<(tw + 255) / 256, 256, 0, stream>>>(w1t, W1, E_, DMODEL, DEXPERT);
        transpose_cvt_kernel<<<(tw + 255) / 256, 256, 0, stream>>>(w2t, W2, E_, DEXPERT, DMODEL);
        n = DFF * DMODEL;
        cvt_f16_kernel<<<(n + 255) / 256, 256, 0, stream>>>(sw1h, SW1, n);
        n = DMODEL * DFF;
        cvt_f16_kernel<<<(n + 255) / 256, 256, 0, stream>>>(sw2h, SW2, n);
    }
    // 2) routing metadata init + router + ordered capacity scan
    init_meta_kernel<<<(E_ * CAP + 255) / 256, 256, 0, stream>>>(stok, ssum, cnts);
    router_kernel<<<NTOK, 256, 0, stream>>>(X, RW, GW, tke, tkg, gsh, ssum, cnts);
    route_scan_kernel<<<1, 1024, 0, stream>>>(tke, tkg, stok, sgate);

    // 3) shared expert: h_sh = gelu(x @ sw1^T) ; y = g_shared * (h_sh @ sw2^T)
    gemm_wmma_kernel<0, false><<<dim3(NTOK / 256, DFF / 64, 1), 128, 0, stream>>>(
        xh, sw1h, hsh, nullptr, nullptr, nullptr,
        DMODEL, DMODEL, DMODEL, DFF, 0, 0, 0, 0);
    gemm_wmma_kernel<1, false><<<dim3(NTOK / 256, DMODEL / 64, 1), 128, 0, stream>>>(
        hsh, sw2h, nullptr, y, nullptr, gsh,
        DFF, DFF, DFF, 0, 0, 0, 0, 0);

    // 4) MoE experts: h = gelu(gather(x) @ w1[e]) ; y += gate * (h @ w2[e])
    gemm_wmma_kernel<0, true><<<dim3(CAP / 256, DEXPERT / 64, E_), 128, 0, stream>>>(
        xh, w1t, hmoe, nullptr, stok, nullptr,
        DMODEL, DMODEL, DMODEL, DEXPERT, CAP, 0, DEXPERT * DMODEL, CAP * DEXPERT);
    gemm_wmma_kernel<2, false><<<dim3(CAP / 256, DMODEL / 64, E_), 128, 0, stream>>>(
        hmoe, w2t, nullptr, y, stok, sgate,
        DEXPERT, DEXPERT, DEXPERT, 0, CAP, CAP * DEXPERT, DMODEL * DEXPERT, 0);

    // 5) stats tail
    stats_kernel<<<1, 32, 0, stream>>>(y + (size_t)NTOK * DMODEL, ssum, cnts);
}
